// lstm_aggregate_55954833932349
// MI455X (gfx1250) — matrix operand order repaired and checked by reading, not compiled, not run on hardware
//
#include <hip/hip_runtime.h>

// ---------------------------------------------------------------------------
// 2-layer LSTM (H=400, T=512, B=64) + batch mean, for MI455X (gfx1250).
//
// Structure:
//   1. Convert weights/inputs to f16, K padded 400 -> 416 (13 x 32 WMMA chunks).
//   2. Big parallel GEMM per layer: XW = x @ W_ih^T + (b_ih + b_hh), f32 out,
//      using v_wmma_f32_16x16x32_f16 (1 wave computes a 16x64 tile).
//   3. 512 sequential step kernels per layer: gates = XW_t + h @ W_hh^T via
//      WMMA, fused LSTM pointwise on the accumulator layout (no LDS needed
//      since one wave owns all four i/f/g/o tiles of its (b,h) 16x16 tile).
//   4. Deterministic batch-mean reduction -> (512, 400) f32.
//
// K-loops are fully unrolled (13 chunks) so accumulators have no loop-carried
// PHIs: no v_mov accumulator copies, no WMMA->VALU hazard NOPs, and the
// scheduler can clause all global_load_b128s ahead of the WMMA stream.
// ---------------------------------------------------------------------------

typedef __attribute__((ext_vector_type(16))) _Float16 v16h;
typedef __attribute__((ext_vector_type(8)))  _Float16 v8h;
typedef __attribute__((ext_vector_type(8)))  float    v8f;

#define LSTM_T   512
#define LSTM_B   64
#define LSTM_H   400
#define LSTM_G4  1600          // 4*H
#define LSTM_KP  416           // H padded to multiple of 32
#define LSTM_TB  (LSTM_T * LSTM_B)  // 32768

// ---------------- fragment loaders (per CDNA5 WMMA VGPR layouts) ------------

// A fragment: 16x32 f16 tile, rows m0..m0+15 of `src` (leading dim ld, f16
// elems), cols k0..k0+31. Lane<16: K in {k0..k0+7, k0+16..k0+23};
// lane>=16: K in {k0+8..k0+15, k0+24..k0+31}. Two 16B loads per lane.
static __device__ __forceinline__ v16h load_frag_a(const _Float16* __restrict__ src,
                                                   int ld, int k0) {
  const int lane = threadIdx.x & 31;
  const int row  = lane & 15;
  const int koff = (lane < 16) ? 0 : 8;
  const _Float16* p = src + (size_t)row * ld + k0 + koff;
  v8h lo = *(const v8h*)(p);
  v8h hi = *(const v8h*)(p + 16);
  v16h r;
#pragma unroll
  for (int i = 0; i < 8; ++i) { r[i] = lo[i]; r[i + 8] = hi[i]; }
  return r;
}

// B fragment: 32x16 f16 tile of W^T, i.e. B[k][n] = W[n0+n][k] with W stored
// row-major (rows = output columns, leading dim ld). Lane<16: col n=lane,
// K in {k0..k0+7, k0+16..k0+23}; lane>=16: col n=lane-16, K in {k0+8..k0+15, k0+24..k0+31}.
static __device__ __forceinline__ v16h load_frag_b(const _Float16* __restrict__ w,
                                                   int ld, int n0, int k0) {
  const int lane = threadIdx.x & 31;
  const int col  = lane & 15;
  const int koff = (lane < 16) ? 0 : 8;
  const _Float16* p = w + (size_t)(n0 + col) * ld + k0 + koff;
  v8h lo = *(const v8h*)(p);
  v8h hi = *(const v8h*)(p + 16);
  v16h r;
#pragma unroll
  for (int i = 0; i < 8; ++i) { r[i] = lo[i]; r[i + 8] = hi[i]; }
  return r;
}

static __device__ __forceinline__ float fast_sigmoid(float x) {
  return 1.0f / (1.0f + __expf(-x));
}
static __device__ __forceinline__ float fast_tanh(float x) {
  // tanh(x) = 1 - 2/(exp(2x)+1); saturates correctly for |x| large.
  return 1.0f - 2.0f / (__expf(2.0f * x) + 1.0f);
}

// ---------------- prep kernels ----------------------------------------------

// f32 (rows x 400) -> f16 (rows x 416), zero-padded cols 400..415
__global__ void cvt_pad_f16_kernel(const float* __restrict__ in,
                                   _Float16* __restrict__ out, int rows) {
  int idx = blockIdx.x * blockDim.x + threadIdx.x;
  int total = rows * LSTM_KP;
  if (idx >= total) return;
  int r = idx / LSTM_KP, c = idx - r * LSTM_KP;
  out[idx] = (c < LSTM_H) ? (_Float16)in[(size_t)r * LSTM_H + c] : (_Float16)0.0f;
}

__global__ void bias_sum_kernel(const float* __restrict__ b0,
                                const float* __restrict__ b1,
                                float* __restrict__ out) {
  int idx = blockIdx.x * blockDim.x + threadIdx.x;
  if (idx < LSTM_G4) out[idx] = b0[idx] + b1[idx];
}

// zero h_cur (64 x 416 f16) and c_cur (64 x 400 f32)
__global__ void init_state_kernel(_Float16* __restrict__ h_cur,
                                  float* __restrict__ c_cur) {
  int idx = blockIdx.x * blockDim.x + threadIdx.x;
  if (idx < LSTM_B * LSTM_KP) h_cur[idx] = (_Float16)0.0f;
  if (idx < LSTM_B * LSTM_H)  c_cur[idx] = 0.0f;
}

// zero the pad columns (400..415) of a (TB x 416) f16 sequence buffer
__global__ void zero_pad_seq_kernel(_Float16* __restrict__ seq) {
  int idx = blockIdx.x * blockDim.x + threadIdx.x;
  int total = LSTM_TB * 16;
  if (idx >= total) return;
  int r = idx >> 4, c = LSTM_H + (idx & 15);
  seq[(size_t)r * LSTM_KP + c] = (_Float16)0.0f;
}

// ---------------- big input-projection GEMM ---------------------------------
// XW[r, g] = sum_k A[r,k] * W[g,k] + bias[g]; one wave -> 16 rows x 64 cols.
__global__ void __launch_bounds__(32)
gemm_xw_kernel(const _Float16* __restrict__ A,    // (TB, KP) f16
               const _Float16* __restrict__ W,    // (G4, KP) f16
               const float*    __restrict__ bias, // (G4)
               float*          __restrict__ XW) { // (TB, G4) f32
  const int r0 = blockIdx.x * 16;
  const int g0 = blockIdx.y * 64;
  v8f acc[4] = {};
  const _Float16* arow = A + (size_t)r0 * LSTM_KP;
#pragma unroll
  for (int k0 = 0; k0 < LSTM_KP; k0 += 32) {
    v16h a = load_frag_a(arow, LSTM_KP, k0);
#pragma unroll
    for (int t = 0; t < 4; ++t) {
      v16h b = load_frag_b(W, LSTM_KP, g0 + t * 16, k0);
      acc[t] = __builtin_amdgcn_wmma_f32_16x16x32_f16(
          false, a, false, b, (short)0, acc[t], false, false);
    }
  }
  const int lane = threadIdx.x & 31;
  const int n  = lane & 15;
  const int mb = (lane < 16) ? 0 : 8;
#pragma unroll
  for (int t = 0; t < 4; ++t) {
    const int g = g0 + t * 16 + n;
    const float bv = bias[g];
#pragma unroll
    for (int j = 0; j < 8; ++j)
      XW[(size_t)(r0 + mb + j) * LSTM_G4 + g] = acc[t][j] + bv;
  }
}

// ---------------- recurrent step --------------------------------------------
// One wave per (16-batch x 16-h) tile; computes all four gate tiles so the
// LSTM pointwise runs directly on the WMMA accumulator layout.
__global__ void __launch_bounds__(32)
lstm_step_kernel(const float*    __restrict__ XWt,   // (64, 1600) for step t
                 const _Float16* __restrict__ Whh,   // (1600, 416) f16
                 _Float16*       __restrict__ h_cur, // (64, 416) f16
                 float*          __restrict__ c_cur, // (64, 400) f32
                 _Float16*       __restrict__ seq)   // (64, 416) output slice
{
  const int b0 = blockIdx.x * 16;   // batch tile
  const int n0 = blockIdx.y * 16;   // hidden-index tile
  v8f acc[4] = {};                  // i, f, g, o
  const _Float16* hrow = h_cur + (size_t)b0 * LSTM_KP;
#pragma unroll
  for (int k0 = 0; k0 < LSTM_KP; k0 += 32) {
    v16h a = load_frag_a(hrow, LSTM_KP, k0);
#pragma unroll
    for (int gt = 0; gt < 4; ++gt) {
      v16h b = load_frag_b(Whh, LSTM_KP, n0 + gt * LSTM_H, k0);
      acc[gt] = __builtin_amdgcn_wmma_f32_16x16x32_f16(
          false, a, false, b, (short)0, acc[gt], false, false);
    }
  }
  const int lane = threadIdx.x & 31;
  const int n  = lane & 15;
  const int mb = (lane < 16) ? 0 : 8;
#pragma unroll
  for (int j = 0; j < 8; ++j) {
    const int b = b0 + mb + j;
    const int h = n0 + n;
    const size_t xo = (size_t)b * LSTM_G4 + h;
    const float gi = acc[0][j] + XWt[xo + 0 * LSTM_H];
    const float gf = acc[1][j] + XWt[xo + 1 * LSTM_H];
    const float gg = acc[2][j] + XWt[xo + 2 * LSTM_H];
    const float go = acc[3][j] + XWt[xo + 3 * LSTM_H];
    const float iv = fast_sigmoid(gi);
    const float fv = fast_sigmoid(gf);
    const float gv = fast_tanh(gg);
    const float ov = fast_sigmoid(go);
    const size_t co = (size_t)b * LSTM_H + h;
    const float cn = fv * c_cur[co] + iv * gv;
    const float hn = ov * fast_tanh(cn);
    c_cur[co] = cn;
    const _Float16 hh = (_Float16)hn;
    const size_t ho = (size_t)b * LSTM_KP + h;
    h_cur[ho] = hh;
    seq[ho]   = hh;
  }
}

// ---------------- batch mean ------------------------------------------------
__global__ void mean_kernel(const _Float16* __restrict__ h2seq, // (TB, 416)
                            float* __restrict__ out) {          // (T, 400)
  int idx = blockIdx.x * blockDim.x + threadIdx.x;
  if (idx >= LSTM_T * LSTM_H) return;
  int t = idx / LSTM_H, h = idx - t * LSTM_H;
  float s = 0.0f;
  const _Float16* base = h2seq + (size_t)t * LSTM_B * LSTM_KP + h;
#pragma unroll 8
  for (int b = 0; b < LSTM_B; ++b) s += (float)base[(size_t)b * LSTM_KP];
  out[idx] = s * (1.0f / LSTM_B);
}

// ---------------- host driver ----------------------------------------------

extern "C" void kernel_launch(void* const* d_in, const int* in_sizes, int n_in,
                              void* d_out, int out_size, void* d_ws, size_t ws_size,
                              hipStream_t stream) {
  const float* clip  = (const float*)d_in[0];
  const float* Wih0  = (const float*)d_in[1];
  const float* Whh0  = (const float*)d_in[2];
  const float* bih0  = (const float*)d_in[3];
  const float* bhh0  = (const float*)d_in[4];
  const float* Wih1  = (const float*)d_in[5];
  const float* Whh1  = (const float*)d_in[6];
  const float* bih1  = (const float*)d_in[7];
  const float* bhh1  = (const float*)d_in[8];
  float* out = (float*)d_out;

  // workspace carve-up (256B aligned); total ~272 MB
  char* ws = (char*)d_ws;
  size_t off = 0;
  auto carve = [&](size_t bytes) -> char* {
    char* p = ws + off;
    off = (off + bytes + 255) & ~(size_t)255;
    return p;
  };
  const size_t wbytes = (size_t)LSTM_G4 * LSTM_KP * sizeof(_Float16);
  _Float16* Wih0h = (_Float16*)carve(wbytes);
  _Float16* Whh0h = (_Float16*)carve(wbytes);
  _Float16* Wih1h = (_Float16*)carve(wbytes);
  _Float16* Whh1h = (_Float16*)carve(wbytes);
  float*    bias0 = (float*)carve(LSTM_G4 * sizeof(float));
  float*    bias1 = (float*)carve(LSTM_G4 * sizeof(float));
  _Float16* xh    = (_Float16*)carve((size_t)LSTM_TB * LSTM_KP * sizeof(_Float16));
  _Float16* seqh  = (_Float16*)carve((size_t)LSTM_TB * LSTM_KP * sizeof(_Float16));
  float*    XW    = (float*)carve((size_t)LSTM_TB * LSTM_G4 * sizeof(float));
  _Float16* hcur  = (_Float16*)carve((size_t)LSTM_B * LSTM_KP * sizeof(_Float16));
  float*    ccur  = (float*)carve((size_t)LSTM_B * LSTM_H * sizeof(float));
  (void)ws_size; (void)in_sizes; (void)n_in; (void)out_size;

  const int TPB = 256;
  auto blocks = [](size_t n, int t) { return (unsigned)((n + t - 1) / t); };

  // 1. precision conversion + bias fusion
  hipLaunchKernelGGL(cvt_pad_f16_kernel, dim3(blocks((size_t)LSTM_G4 * LSTM_KP, TPB)), dim3(TPB), 0, stream, Wih0, Wih0h, LSTM_G4);
  hipLaunchKernelGGL(cvt_pad_f16_kernel, dim3(blocks((size_t)LSTM_G4 * LSTM_KP, TPB)), dim3(TPB), 0, stream, Whh0, Whh0h, LSTM_G4);
  hipLaunchKernelGGL(cvt_pad_f16_kernel, dim3(blocks((size_t)LSTM_G4 * LSTM_KP, TPB)), dim3(TPB), 0, stream, Wih1, Wih1h, LSTM_G4);
  hipLaunchKernelGGL(cvt_pad_f16_kernel, dim3(blocks((size_t)LSTM_G4 * LSTM_KP, TPB)), dim3(TPB), 0, stream, Whh1, Whh1h, LSTM_G4);
  hipLaunchKernelGGL(bias_sum_kernel, dim3(blocks(LSTM_G4, TPB)), dim3(TPB), 0, stream, bih0, bhh0, bias0);
  hipLaunchKernelGGL(bias_sum_kernel, dim3(blocks(LSTM_G4, TPB)), dim3(TPB), 0, stream, bih1, bhh1, bias1);
  hipLaunchKernelGGL(cvt_pad_f16_kernel, dim3(blocks((size_t)LSTM_TB * LSTM_KP, TPB)), dim3(TPB), 0, stream, clip, xh, LSTM_TB);
  hipLaunchKernelGGL(zero_pad_seq_kernel, dim3(blocks((size_t)LSTM_TB * 16, TPB)), dim3(TPB), 0, stream, seqh);

  const dim3 gemmGrid(LSTM_TB / 16, LSTM_G4 / 64);  // (2048, 25)
  const dim3 stepGrid(LSTM_B / 16, LSTM_H / 16);    // (4, 25)
  const dim3 wave(32);

  // ---- layer 0 ----
  hipLaunchKernelGGL(init_state_kernel, dim3(blocks(LSTM_B * LSTM_KP, TPB)), dim3(TPB), 0, stream, hcur, ccur);
  hipLaunchKernelGGL(gemm_xw_kernel, gemmGrid, wave, 0, stream, xh, Wih0h, bias0, XW);
  for (int t = 0; t < LSTM_T; ++t) {
    hipLaunchKernelGGL(lstm_step_kernel, stepGrid, wave, 0, stream,
                       XW + (size_t)t * LSTM_B * LSTM_G4, Whh0h, hcur, ccur,
                       seqh + (size_t)t * LSTM_B * LSTM_KP);
  }

  // ---- layer 1 (reuse XW; recycle xh as h2 sequence buffer) ----
  hipLaunchKernelGGL(init_state_kernel, dim3(blocks(LSTM_B * LSTM_KP, TPB)), dim3(TPB), 0, stream, hcur, ccur);
  hipLaunchKernelGGL(gemm_xw_kernel, gemmGrid, wave, 0, stream, seqh, Wih1h, bias1, XW);
  for (int t = 0; t < LSTM_T; ++t) {
    hipLaunchKernelGGL(lstm_step_kernel, stepGrid, wave, 0, stream,
                       XW + (size_t)t * LSTM_B * LSTM_G4, Whh1h, hcur, ccur,
                       xh + (size_t)t * LSTM_B * LSTM_KP);
  }

  // ---- batch mean -> (512, 400) ----
  hipLaunchKernelGGL(mean_kernel, dim3(blocks((size_t)LSTM_T * LSTM_H, TPB)), dim3(TPB), 0, stream, xh, out);
}
